// AdditiveAttention_3143916061080
// MI455X (gfx1250) — compile-verified, matrix operand order checked by reading, not run on hardware
//
#include <hip/hip_runtime.h>
#include <hip/hip_bf16.h>
#include <math.h>

typedef __attribute__((ext_vector_type(16))) __bf16 v16bf;
typedef __attribute__((ext_vector_type(4)))  __bf16 v4bf;
typedef __attribute__((ext_vector_type(8)))  float  v8f;
typedef __attribute__((ext_vector_type(4)))  float  v4f;

static constexpr int N_ = 32, L_ = 2048, H_ = 1024, A_ = 1024;
static constexpr int PITCH = 1032;   // bf16 elems; 2064B row stride -> spreads rows across all 64 banks
static constexpr size_t SMEM_BYTES = (size_t)4 * 32 * PITCH * 2 + 2 * A_ * sizeof(float);

union FragU { v16bf v; uint4 q[2]; };

#define LDFRAG(dst, base, kpos)                                   \
    do {                                                          \
        (dst).q[0] = *(const uint4*)((base) + (kpos));            \
        (dst).q[1] = *(const uint4*)((base) + (kpos) + 16);       \
    } while (0)

#define WMMA_BF16(A, B, C)                                        \
    __builtin_amdgcn_wmma_f32_16x16x32_bf16(false, (A).v, false, (B).v, (short)0, (C), false, false)

__device__ __forceinline__ float fast_tanh(float x) {
#if __has_builtin(__builtin_amdgcn_tanhf)
    return __builtin_amdgcn_tanhf(x);      // single V_TANH_F32 (TRANS, co-executes with XDL)
#else
    return tanhf(x);
#endif
}

// ---------------- kernel 0: W_k fp32 -> bf16 (stays hot in 192MB L2) ----------------
__global__ void k_cvt_wk(const float* __restrict__ wk, __bf16* __restrict__ out, int n) {
    int i = blockIdx.x * 256 + threadIdx.x;
    if (i < n) out[i] = (__bf16)wk[i];
}

// ---------------- kernel 1: q = query @ W_q^T  [N,A] ----------------
__global__ void k_proj_q(const float* __restrict__ query, const float* __restrict__ Wq,
                         float* __restrict__ q) {
    __shared__ float qs[H_];
    int flat = blockIdx.x * 256 + threadIdx.x;       // one n per block (256 | 1024)
    int n = flat >> 10, a = flat & (A_ - 1);
    for (int i = threadIdx.x; i < H_; i += 256) qs[i] = query[n * H_ + i];
    __syncthreads();
    const float* wrow = Wq + (size_t)a * H_;
    float acc = 0.f;
    for (int h = 0; h < H_; h += 4) {
        float4 w = *(const float4*)(wrow + h);
        acc += w.x * qs[h] + w.y * qs[h + 1] + w.z * qs[h + 2] + w.w * qs[h + 3];
    }
    q[flat] = acc;
}

// ---------------- kernel 2: fused scores, 2x2 register-tiled bf16 WMMA ----------------
// grid: (L/128, N); 128 threads = 4 waves; wave w owns 32 l-rows (two 16-row M tiles)
__global__ void __launch_bounds__(128, 1)
k_scores(const float* __restrict__ keys, const __bf16* __restrict__ wk,
         const float* __restrict__ q, const float* __restrict__ v,
         float* __restrict__ scores) {
    extern __shared__ char smem[];
    const int tid   = threadIdx.x;
    const int wave  = tid >> 5;            // 0..3
    const int lane  = tid & 31;
    const int idx16 = lane & 15;
    const int half  = lane >> 4;
    const int koff  = half * 8;            // lanes 0-15: K{0..7,16..23}; lanes 16-31: K{8..15,24..31}
    const int n  = blockIdx.y;
    const int l0 = blockIdx.x * 128 + wave * 32;

    __bf16* alds = (__bf16*)smem + (size_t)wave * 32 * PITCH;
    float*  qlds = (float*)(smem + (size_t)4 * 32 * PITCH * 2);
    float*  vlds = qlds + A_;

    // stage 32 keys rows (fp32 -> bf16, non-temporal: don't evict W_k from L2)
    const float* ktile = keys + ((size_t)n * L_ + l0) * H_;
    for (int e = lane * 4; e < 32 * H_; e += 32 * 4) {
        v4f f = __builtin_nontemporal_load((const v4f*)(ktile + e));
        int r = e >> 10, c = e & (H_ - 1);
        v4bf pk = { (__bf16)f.x, (__bf16)f.y, (__bf16)f.z, (__bf16)f.w };
        *(v4bf*)(alds + (size_t)r * PITCH + c) = pk;
    }
    for (int i = tid; i < A_; i += 128) { qlds[i] = q[n * A_ + i]; vlds[i] = v[i]; }
    __syncthreads();

    const __bf16* arow0b = alds + (size_t)idx16 * PITCH + koff;  // M tile 0: rows 0..15
    const __bf16* arow1b = arow0b + 16 * PITCH;                  // M tile 1: rows 16..31

    float sacc0[8] = {0,0,0,0,0,0,0,0};
    float sacc1[8] = {0,0,0,0,0,0,0,0};

    for (int at = 0; at < A_; at += 32) {
        // Opaque zero offset: stops LICM from hoisting the (at-invariant) A-fragment
        // LDS loads out of this loop and spilling them to scratch. Keeps them as
        // per-iteration ds_load_b128 on the (otherwise idle) LDS pipe.
        int kz = 0;
        asm volatile("" : "+v"(kz));
        const __bf16* arow0 = arow0b + kz;
        const __bf16* arow1 = arow1b + kz;

        const __bf16* brow0 = wk + (size_t)(at + idx16) * H_ + koff;        // a cols at..+15
        const __bf16* brow1 = brow0 + (size_t)16 * H_;                      // a cols at+16..+31
        v8f acc00 = {0,0,0,0,0,0,0,0}, acc01 = {0,0,0,0,0,0,0,0};
        v8f acc10 = {0,0,0,0,0,0,0,0}, acc11 = {0,0,0,0,0,0,0,0};

        FragU a0x, a1x, b0x, b1x, a0y, a1y, b0y, b1y;
        LDFRAG(a0x, arow0, 0); LDFRAG(a1x, arow1, 0);
        LDFRAG(b0x, brow0, 0); LDFRAG(b1x, brow1, 0);

#pragma unroll
        for (int k = 0; k < H_; k += 64) {
            // prefetch k+32 set while WMMAs consume the k set
            LDFRAG(a0y, arow0, k + 32); LDFRAG(a1y, arow1, k + 32);
            LDFRAG(b0y, brow0, k + 32); LDFRAG(b1y, brow1, k + 32);
            acc00 = WMMA_BF16(a0x, b0x, acc00);
            acc01 = WMMA_BF16(a0x, b1x, acc01);
            acc10 = WMMA_BF16(a1x, b0x, acc10);
            acc11 = WMMA_BF16(a1x, b1x, acc11);
            // prefetch k+64 (overreads on last iter are never consumed; buffers padded)
            LDFRAG(a0x, arow0, k + 64); LDFRAG(a1x, arow1, k + 64);
            LDFRAG(b0x, brow0, k + 64); LDFRAG(b1x, brow1, k + 64);
            acc00 = WMMA_BF16(a0y, b0y, acc00);
            acc01 = WMMA_BF16(a0y, b1y, acc01);
            acc10 = WMMA_BF16(a1y, b0y, acc10);
            acc11 = WMMA_BF16(a1y, b1y, acc11);
        }

        // epilogue: bias + tanh + weight-by-v, accumulate per-row partials
        float q0 = qlds[at + idx16],      v0 = vlds[at + idx16];
        float q1 = qlds[at + 16 + idx16], v1 = vlds[at + 16 + idx16];
#pragma unroll
        for (int r = 0; r < 8; ++r) {
            sacc0[r] += fast_tanh(acc00[r] + q0) * v0 + fast_tanh(acc01[r] + q1) * v1;
            sacc1[r] += fast_tanh(acc10[r] + q0) * v0 + fast_tanh(acc11[r] + q1) * v1;
        }
    }

    // reduce over the 16 a-columns held across lanes of each half-wave
#pragma unroll
    for (int r = 0; r < 8; ++r) {
        float s0 = sacc0[r], s1 = sacc1[r];
        s0 += __shfl_xor(s0, 1, 32); s1 += __shfl_xor(s1, 1, 32);
        s0 += __shfl_xor(s0, 2, 32); s1 += __shfl_xor(s1, 2, 32);
        s0 += __shfl_xor(s0, 4, 32); s1 += __shfl_xor(s1, 4, 32);
        s0 += __shfl_xor(s0, 8, 32); s1 += __shfl_xor(s1, 8, 32);
        sacc0[r] = s0; sacc1[r] = s1;
    }
    if (idx16 == 0) {
#pragma unroll
        for (int r = 0; r < 8; ++r) {
            int m = half * 8 + r;   // C/D layout: VGPR r = row r (lanes 0-15) / row r+8 (lanes 16-31)
            scores[(size_t)n * L_ + l0 + m]      = sacc0[r];
            scores[(size_t)n * L_ + l0 + 16 + m] = sacc1[r];
        }
    }
}

// ---------------- kernel 3: masked softmax over L per batch row ----------------
__global__ void k_softmax(const float* __restrict__ scores, const unsigned char* __restrict__ mask,
                          float* __restrict__ attn) {
    __shared__ float red[256];
    int n = blockIdx.x, tid = threadIdx.x;
    const float* srow = scores + (size_t)n * L_;
    const unsigned char* mrow = mask + (size_t)n * L_;
    float lmax = -INFINITY;
    for (int l = tid; l < L_; l += 256) {
        float s = mrow[l] ? srow[l] : -INFINITY;
        lmax = fmaxf(lmax, s);
    }
    red[tid] = lmax; __syncthreads();
    for (int off = 128; off; off >>= 1) { if (tid < off) red[tid] = fmaxf(red[tid], red[tid + off]); __syncthreads(); }
    float m = red[0]; __syncthreads();
    float lsum = 0.f;
    for (int l = tid; l < L_; l += 256) {
        float s = mrow[l] ? srow[l] : -INFINITY;
        float e = expf(s - m);
        attn[(size_t)n * L_ + l] = e;
        lsum += e;
    }
    red[tid] = lsum; __syncthreads();
    for (int off = 128; off; off >>= 1) { if (tid < off) red[tid] += red[tid + off]; __syncthreads(); }
    float inv = 1.0f / red[0];
    for (int l = tid; l < L_; l += 256) attn[(size_t)n * L_ + l] *= inv;
}

// ---------------- kernel 4: context[n,h] = sum_l attn[n,l]*keys[n,l,h] (HBM-bound stream) ----
__global__ void k_context(const float* __restrict__ attn, const float* __restrict__ keys,
                          float* __restrict__ ctx) {
    __shared__ float w[L_];
    int n = blockIdx.y;
    int h = blockIdx.x * 256 + threadIdx.x;
    for (int l = threadIdx.x; l < L_; l += 256) w[l] = attn[(size_t)n * L_ + l];
    __syncthreads();
    const float* kb = keys + (size_t)n * L_ * H_ + h;
    float acc = 0.f;
    for (int l = 0; l < L_; ++l) acc += w[l] * __builtin_nontemporal_load(kb + (size_t)l * H_);
    ctx[(size_t)n * H_ + h] = acc;
}

extern "C" void kernel_launch(void* const* d_in, const int* in_sizes, int n_in,
                              void* d_out, int out_size, void* d_ws, size_t ws_size,
                              hipStream_t stream) {
    const float* query = (const float*)d_in[0];
    const float* keys  = (const float*)d_in[1];
    const unsigned char* mask = (const unsigned char*)d_in[2];
    const float* Wq = (const float*)d_in[3];
    const float* Wk = (const float*)d_in[4];
    const float* v  = (const float*)d_in[5];

    float* ctx  = (float*)d_out;            // [N,H]
    float* attn = ctx + N_ * H_;            // [N,L]

    char* ws = (char*)d_ws;
    float*  qws    = (float*)ws;                                          // N*A f32   (128 KB)
    __bf16* wkbf   = (__bf16*)(ws + (size_t)N_ * A_ * sizeof(float));     // A*H bf16  (2 MB)
    // 4 KB pad after wkbf absorbs prefetch overreads from k_scores
    float*  scores = (float*)(ws + (size_t)N_ * A_ * sizeof(float)
                                 + (size_t)A_ * H_ * sizeof(__bf16) + 4096);  // N*L f32

    k_cvt_wk <<<(A_ * H_) / 256, 256, 0, stream>>>(Wk, wkbf, A_ * H_);
    k_proj_q <<<(N_ * A_) / 256, 256, 0, stream>>>(query, Wq, qws);
    dim3 g2(L_ / 128, N_);
    k_scores <<<g2, 128, SMEM_BYTES, stream>>>(keys, wkbf, qws, v, scores);
    k_softmax<<<N_, 256, 0, stream>>>(scores, mask, attn);
    k_context<<<dim3(H_ / 256, N_), 256, 0, stream>>>(attn, keys, ctx);
}
